// RPN_19937238188158
// MI455X (gfx1250) — matrix operand order repaired and checked by reading, not compiled, not run on hardware
//
#include <hip/hip_runtime.h>
#include <hip/hip_bf16.h>
#include <math.h>

typedef __attribute__((ext_vector_type(16))) __bf16 v16bf;
typedef __attribute__((ext_vector_type(8)))  __bf16 v8bf;
typedef __attribute__((ext_vector_type(8)))  float  v8f;
typedef __attribute__((ext_vector_type(4)))  int    v4i;

#define HFE 64
#define WFE 64
#define NB 4
#define CIN 768
#define CMID 512
#define KTOT (CIN * 9)        // 6912
#define NPIX (HFE * WFE)      // 4096
#define MTOT (NB * NPIX)      // 16384
#define NANCH 12
#define NA_TOT (NPIX * NANCH) // 49152
#define SORTN 65536
#define PRE_NMS_N 6000
#define POST_NMS_N 300
#define NMS_TH 0.7f

// ---------------------------------------------------------------------------
// CDNA5 async global->LDS copy (ASYNCcnt-tracked), with fallback
// Builtin signature (from compiler diagnostic): (v4i addrspace(1)*, v4i addrspace(3)*, imm, imm)
// ---------------------------------------------------------------------------
#if __has_builtin(__builtin_amdgcn_global_load_async_to_lds_b128)
#define HAVE_ASYNC_LDS 1
__device__ __forceinline__ void async_copy_b128(const void* g, void* l) {
    __attribute__((address_space(1))) void* gv =
        (__attribute__((address_space(1))) void*)const_cast<void*>(g);
    __attribute__((address_space(3))) void* lv =
        (__attribute__((address_space(3))) void*)l;
    __builtin_amdgcn_global_load_async_to_lds_b128(
        (__attribute__((address_space(1))) v4i*)gv,
        (__attribute__((address_space(3))) v4i*)lv,
        0, 0);
}
__device__ __forceinline__ void wait_async0() {
#if __has_builtin(__builtin_amdgcn_s_wait_asynccnt)
    __builtin_amdgcn_s_wait_asynccnt(0);
#else
    asm volatile("s_wait_asynccnt 0x0" ::: "memory");
#endif
}
#else
#define HAVE_ASYNC_LDS 0
__device__ __forceinline__ void wait_async0() {}
#endif

// ---------------------------------------------------------------------------
// Prep kernels: fp32 -> bf16 layout changes
// ---------------------------------------------------------------------------
__global__ void prep_features_nhwc(const float* __restrict__ F, __bf16* __restrict__ Fb) {
    size_t tid = (size_t)blockIdx.x * blockDim.x + threadIdx.x;
    if (tid >= (size_t)NB * NPIX * CIN) return;
    int c = (int)(tid % CIN);
    size_t p = tid / CIN;                 // b*4096 + y*64 + x
    int b = (int)(p >> 12);
    int y = (int)((p >> 6) & 63);
    int x = (int)(p & 63);
    Fb[tid] = (__bf16)F[(((size_t)b * CIN + c) * HFE + y) * WFE + x];
}

__global__ void prep_weights_km(const float* __restrict__ W1, __bf16* __restrict__ Wt) {
    size_t tid = (size_t)blockIdx.x * blockDim.x + threadIdx.x;
    if (tid >= (size_t)CMID * KTOT) return;
    int n = (int)(tid / KTOT);
    int k = (int)(tid % KTOT);
    int tap = k / CIN;                    // 0..8  (ky*3+kx)
    int ci  = k % CIN;
    Wt[tid] = (__bf16)W1[(((size_t)n * CIN + ci) * 3 + (tap / 3)) * 3 + (tap % 3)];
}

__global__ void prep_head_weights(const float* __restrict__ Ws, const float* __restrict__ Wb,
                                  __bf16* __restrict__ Wh) {
    int tid = blockIdx.x * blockDim.x + threadIdx.x;
    if (tid >= 80 * CMID) return;
    int n = tid / CMID;
    int k = tid % CMID;
    float v = 0.f;
    if (n < 24)       v = Ws[n * CMID + k];
    else if (n < 72)  v = Wb[(n - 24) * CMID + k];
    Wh[tid] = (__bf16)v;
}

// ---------------------------------------------------------------------------
// Conv 3x3 (768->512) + bias + ReLU as implicit GEMM with bf16 WMMA
// M=16384 (pixels), N=512, K=6912.  Tile 128x128, Kchunk=32, double-buffer LDS
// Staging uses GLOBAL_LOAD_ASYNC_TO_LDS_B128 (ASYNCcnt) when available.
// ---------------------------------------------------------------------------
__global__ __launch_bounds__(256)
void conv3x3_wmma(const __bf16* __restrict__ F, const __bf16* __restrict__ Wt,
                  const float* __restrict__ bias, __bf16* __restrict__ X) {
    __shared__ __bf16 As[2][128 * 32];   // [m][k] row-major, k fastest
    __shared__ __bf16 Bs[2][128 * 32];   // [n][k] row-major, k fastest

    const int t    = threadIdx.x;
    const int lane = t & 31;
    const int wid  = t >> 5;             // 8 waves
    const int wm   = (wid & 1) * 64;     // wave M offset within tile
    const int wn   = (wid >> 1) * 32;    // wave N offset within tile
    const int m_base = blockIdx.y * 128;
    const int n_base = blockIdx.x * 128;

    // staging: thread t handles 16 bf16 (32B) of one row
    const int srow  = t >> 1;            // 0..127
    const int shalf = (t & 1) * 16;      // k offset 0 or 16

    const int pix = m_base + srow;       // global pixel for A staging
    const int pb  = pix >> 12;
    const int py  = (pix >> 6) & 63;
    const int px  = pix & 63;

    const v8f vz = {0.f,0.f,0.f,0.f,0.f,0.f,0.f,0.f};
    v8f acc[4][2];
    #pragma unroll
    for (int i = 0; i < 4; ++i)
        #pragma unroll
        for (int j = 0; j < 2; ++j) acc[i][j] = vz;

    const int NCH = KTOT / 32;           // 216 K-chunks

    auto stage = [&](int c, int buf) {
        // ---- A tile (im2col on the fly; chunk lies inside one 3x3 tap) ----
        int tap = c / 24;                // 24 chunks of 32 per tap (768/32)
        int ci0 = (c % 24) * 32 + shalf;
        int yy = py + tap / 3 - 1;
        int xx = px + (tap % 3) - 1;
        __bf16* la = &As[buf][srow * 32 + shalf];
        if (yy >= 0 && yy < HFE && xx >= 0 && xx < WFE) {
            const __bf16* src = F + (((size_t)(pb * HFE + yy) * WFE + xx) * CIN + ci0);
#if HAVE_ASYNC_LDS
            async_copy_b128(src, la);
            async_copy_b128(src + 8, la + 8);
#else
            const uint4* p = (const uint4*)src;
            uint4* da = (uint4*)la;
            da[0] = p[0]; da[1] = p[1];
#endif
        } else {
            const uint4 z = {0u,0u,0u,0u};
            uint4* da = (uint4*)la;
            da[0] = z; da[1] = z;        // zero-fill lanes are disjoint from async lanes
        }
        // ---- B tile (weights already [n][k]) ----
        const __bf16* wsrc = Wt + ((size_t)(n_base + srow) * KTOT + c * 32 + shalf);
        __bf16* lb = &Bs[buf][srow * 32 + shalf];
#if HAVE_ASYNC_LDS
        async_copy_b128(wsrc, lb);
        async_copy_b128(wsrc + 8, lb + 8);
#else
        const uint4* q = (const uint4*)wsrc;
        uint4* db = (uint4*)lb;
        db[0] = q[0]; db[1] = q[1];
#endif
    };

    stage(0, 0);
    wait_async0();
    __syncthreads();

    const int kb = (lane >> 4) * 8;      // A: K base per lane half
    const int kg = (lane >> 4) * 16;     // unused; B loads use kb, same as A

    for (int c = 0; c < NCH; ++c) {
        const int cb = c & 1;
        if (c + 1 < NCH) stage(c + 1, (c + 1) & 1);

        v16bf af[4], bfr[2];
        #pragma unroll
        for (int ms = 0; ms < 4; ++ms) {
            int m = wm + ms * 16 + (lane & 15);
            ((v8bf*)&af[ms])[0] = *(const v8bf*)&As[cb][m * 32 + kb];        // K = kb..kb+7
            ((v8bf*)&af[ms])[1] = *(const v8bf*)&As[cb][m * 32 + kb + 16];   // K = kb+16..kb+23
        }
        #pragma unroll
        for (int ns = 0; ns < 2; ++ns) {
            int n = wn + ns * 16 + (lane & 15);
            ((v8bf*)&bfr[ns])[0] = *(const v8bf*)&Bs[cb][n * 32 + kb];       // K = kb..kb+7
            ((v8bf*)&bfr[ns])[1] = *(const v8bf*)&Bs[cb][n * 32 + kb + 16];  // K = kb+16..kb+23
        }
        #pragma unroll
        for (int ms = 0; ms < 4; ++ms)
            #pragma unroll
            for (int ns = 0; ns < 2; ++ns)
                acc[ms][ns] = __builtin_amdgcn_wmma_f32_16x16x32_bf16(
                    false, af[ms], false, bfr[ns], (short)0, acc[ms][ns], false, false);
        wait_async0();                   // next chunk's async copies done
        __syncthreads();
    }

    // epilogue: bias + ReLU, store bf16 NHWC activations
    #pragma unroll
    for (int ms = 0; ms < 4; ++ms)
        #pragma unroll
        for (int ns = 0; ns < 2; ++ns)
            #pragma unroll
            for (int r = 0; r < 8; ++r) {
                int M = m_base + wm + ms * 16 + ((lane >> 4) * 8) + r;
                int N = n_base + wn + ns * 16 + (lane & 15);
                float v = acc[ms][ns][r] + bias[N];
                v = fmaxf(v, 0.f);
                X[(size_t)M * CMID + N] = (__bf16)v;
            }
}

// ---------------------------------------------------------------------------
// Fused cls+bbox 1x1 convs: GEMM M=16384, N=80(72 real), K=512, WMMA bf16
// ---------------------------------------------------------------------------
__global__ __launch_bounds__(256)
void heads_wmma(const __bf16* __restrict__ X, const __bf16* __restrict__ Wh,
                float* __restrict__ Head) {
    const int t = threadIdx.x, lane = t & 31, wid = t >> 5;
    const int m_base = blockIdx.x * 128 + wid * 16;
    const v8f vz = {0.f,0.f,0.f,0.f,0.f,0.f,0.f,0.f};
    v8f acc[5];
    #pragma unroll
    for (int i = 0; i < 5; ++i) acc[i] = vz;

    const int kb = (lane >> 4) * 8;
    const int kg = (lane >> 4) * 16;
    const int arow = m_base + (lane & 15);

    for (int kk = 0; kk < 16; ++kk) {
        int k0 = kk * 32;
        v16bf a;
        ((v8bf*)&a)[0] = *(const v8bf*)(X + (size_t)arow * CMID + k0 + kb);
        ((v8bf*)&a)[1] = *(const v8bf*)(X + (size_t)arow * CMID + k0 + kb + 16);
        #pragma unroll
        for (int ns = 0; ns < 5; ++ns) {
            int n = ns * 16 + (lane & 15);
            v16bf b;
            ((v8bf*)&b)[0] = *(const v8bf*)(Wh + (size_t)n * CMID + k0 + kb);
            ((v8bf*)&b)[1] = *(const v8bf*)(Wh + (size_t)n * CMID + k0 + kb + 16);
            acc[ns] = __builtin_amdgcn_wmma_f32_16x16x32_bf16(
                false, a, false, b, (short)0, acc[ns], false, false);
        }
    }
    #pragma unroll
    for (int ns = 0; ns < 5; ++ns)
        #pragma unroll
        for (int r = 0; r < 8; ++r) {
            int M  = m_base + ((lane >> 4) * 8) + r;
            int ch = ns * 16 + (lane & 15);
            Head[(size_t)M * 80 + ch] = acc[ns][r];
        }
}

// ---------------------------------------------------------------------------
// Score (2-class softmax), anchor decode, clip, min-size filter, sort keys
// ---------------------------------------------------------------------------
__global__ void decode_kernel(const float* __restrict__ Head,
                              const float* __restrict__ bs, const float* __restrict__ bb,
                              const float* __restrict__ im_info,
                              float* __restrict__ Boxes,
                              unsigned long long* __restrict__ Keys) {
    int tid = blockIdx.x * blockDim.x + threadIdx.x;
    if (tid >= NB * SORTN) return;
    int b  = tid >> 16;
    int ai = tid & (SORTN - 1);
    unsigned long long* kout = Keys + (size_t)b * SORTN + ai;
    if (ai >= NA_TOT) { *kout = 0ULL; return; }  // padding sorts last

    int pixi = ai / NANCH;
    int a    = ai - pixi * NANCH;
    int y = pixi >> 6, x = pixi & 63;
    size_t gp = (size_t)b * NPIX + pixi;

    float s0 = Head[gp * 80 + a]      + bs[a];
    float s1 = Head[gp * 80 + 12 + a] + bs[12 + a];
    float score = 1.0f / (1.0f + expf(s0 - s1));   // softmax fg prob

    float dx = Head[gp * 80 + 24 + a * 4 + 0] + bb[a * 4 + 0];
    float dy = Head[gp * 80 + 24 + a * 4 + 1] + bb[a * 4 + 1];
    float dw = Head[gp * 80 + 24 + a * 4 + 2] + bb[a * 4 + 2];
    float dh = Head[gp * 80 + 24 + a * 4 + 3] + bb[a * 4 + 3];

    // regenerate base anchor a (ratio outer, scale inner); rintf == numpy round
    const float ratios[3] = {0.5f, 1.0f, 2.0f};
    const float scales[4] = {4.f, 8.f, 16.f, 32.f};
    float r = ratios[a >> 2], s = scales[a & 3];
    float ws0 = rintf(sqrtf(256.0f / r));
    float hs0 = rintf(ws0 * r);
    float w = ws0 * s, h = hs0 * s;
    float ax1 = x * 16 + 7.5f - 0.5f * (w - 1.0f);
    float ay1 = y * 16 + 7.5f - 0.5f * (h - 1.0f);
    float cx = ax1 + 0.5f * w;
    float cy = ay1 + 0.5f * h;

    float pcx = dx * w + cx, pcy = dy * h + cy;
    float pw = expf(dw) * w, ph = expf(dh) * h;

    float imh = im_info[b * 3 + 0], imw = im_info[b * 3 + 1], imsc = im_info[b * 3 + 2];
    float x1 = fminf(fmaxf(pcx - 0.5f * pw, 0.f), imw - 1.f);
    float y1 = fminf(fmaxf(pcy - 0.5f * ph, 0.f), imh - 1.f);
    float x2 = fminf(fmaxf(pcx + 0.5f * pw, 0.f), imw - 1.f);
    float y2 = fminf(fmaxf(pcy + 0.5f * ph, 0.f), imh - 1.f);

    float msz = 16.0f * imsc;
    bool valid = (x2 - x1 + 1.f >= msz) && (y2 - y1 + 1.f >= msz);
    float sc = valid ? score : -INFINITY;

    float* bo = Boxes + ((size_t)b * NA_TOT + ai) * 4;
    bo[0] = x1; bo[1] = y1; bo[2] = x2; bo[3] = y2;

    unsigned u = __float_as_uint(sc);
    u = (u & 0x80000000u) ? ~u : (u | 0x80000000u);   // monotonic float key
    *kout = ((unsigned long long)u << 32) | (unsigned long long)(0xFFFFFFFFu - (unsigned)ai);
}

// ---------------------------------------------------------------------------
// Bitonic sort (descending), one step per launch; 4 independent 65536 segments
// ---------------------------------------------------------------------------
__global__ void bitonic_step(unsigned long long* __restrict__ Keys, int k, int j) {
    int tid = blockIdx.x * blockDim.x + threadIdx.x;
    if (tid >= NB * SORTN) return;
    int b = tid >> 16;
    int i = tid & (SORTN - 1);
    int ixj = i ^ j;
    if (ixj <= i) return;
    unsigned long long* base = Keys + (size_t)b * SORTN;
    unsigned long long a = base[i], c = base[ixj];
    bool up = ((i & k) == 0);
    bool sw = up ? (a < c) : (a > c);   // descending overall
    if (sw) { base[i] = c; base[ixj] = a; }
}

__global__ void gather_sorted(const unsigned long long* __restrict__ Keys,
                              const float* __restrict__ Boxes,
                              float* __restrict__ SB) {
    int tid = blockIdx.x * blockDim.x + threadIdx.x;
    if (tid >= NB * PRE_NMS_N) return;
    int b = tid / PRE_NMS_N, r = tid % PRE_NMS_N;
    unsigned long long key = Keys[(size_t)b * SORTN + r];
    unsigned idx = 0xFFFFFFFFu - (unsigned)(key & 0xFFFFFFFFull);
    float v0 = 0.f, v1 = 0.f, v2 = 0.f, v3 = 0.f;
    if (idx < NA_TOT) {
        const float* bo = Boxes + ((size_t)b * NA_TOT + idx) * 4;
        v0 = bo[0]; v1 = bo[1]; v2 = bo[2]; v3 = bo[3];
    }
    float* o = SB + (size_t)tid * 4;
    o[0] = v0; o[1] = v1; o[2] = v2; o[3] = v3;
}

// ---------------------------------------------------------------------------
// Greedy NMS, 1 workgroup/batch, boxes cached in registers, keep flags in LDS
// Early exit at 300 kept (suppression only flows forward -> output identical)
// ---------------------------------------------------------------------------
__global__ __launch_bounds__(1024)
void nms_kernel(const float* __restrict__ SB, float* __restrict__ out) {
    const int b = blockIdx.x, t = threadIdx.x;
    const float* BX = SB + (size_t)b * PRE_NMS_N * 4;

    float bx1[6], by1[6], bx2[6], by2[6], bar[6];
    #pragma unroll
    for (int k = 0; k < 6; ++k) {
        int j = t + k * 1024;
        if (j < PRE_NMS_N) {
            bx1[k] = BX[j * 4 + 0]; by1[k] = BX[j * 4 + 1];
            bx2[k] = BX[j * 4 + 2]; by2[k] = BX[j * 4 + 3];
            bar[k] = (bx2[k] - bx1[k] + 1.f) * (by2[k] - by1[k] + 1.f);
        } else { bx1[k] = by1[k] = bx2[k] = by2[k] = 0.f; bar[k] = 1.f; }
    }

    __shared__ unsigned char keep[PRE_NMS_N];
    __shared__ float ref[5];
    __shared__ int cnt, flag, brk;
    for (int j = t; j < PRE_NMS_N; j += 1024) keep[j] = 1;
    if (t == 0) { cnt = 0; brk = 0; }
    __syncthreads();

    for (int i = 0; i < PRE_NMS_N; ++i) {
        int owner = i & 1023, ko = i >> 10;
        if (t == owner) {
            if (keep[i]) {
                ref[0] = bx1[ko]; ref[1] = by1[ko]; ref[2] = bx2[ko]; ref[3] = by2[ko];
                ref[4] = bar[ko];
                flag = 1;
                if (++cnt >= POST_NMS_N) brk = 1;
            } else flag = 0;
        }
        __syncthreads();
        if (brk) break;
        if (flag) {
            float rx1 = ref[0], ry1 = ref[1], rx2 = ref[2], ry2 = ref[3], rar = ref[4];
            #pragma unroll
            for (int k = 0; k < 6; ++k) {
                int j = t + k * 1024;
                if (j > i && j < PRE_NMS_N && keep[j]) {
                    float xx1 = fmaxf(rx1, bx1[k]), yy1 = fmaxf(ry1, by1[k]);
                    float xx2 = fminf(rx2, bx2[k]), yy2 = fminf(ry2, by2[k]);
                    float inter = fmaxf(xx2 - xx1 + 1.f, 0.f) * fmaxf(yy2 - yy1 + 1.f, 0.f);
                    float iou = inter / (rar + bar[k] - inter);
                    if (iou > NMS_TH) keep[j] = 0;
                }
            }
        }
        __syncthreads();
    }
    __syncthreads();

    if (t == 0) {
        int r = 0;
        for (int i = 0; i < PRE_NMS_N && r < POST_NMS_N; ++i) {
            if (keep[i]) {
                float* o = out + ((size_t)b * POST_NMS_N + r) * 5;
                o[0] = (float)b;
                o[1] = BX[i * 4 + 0]; o[2] = BX[i * 4 + 1];
                o[3] = BX[i * 4 + 2]; o[4] = BX[i * 4 + 3];
                ++r;
            }
        }
        for (; r < POST_NMS_N; ++r) {   // padded rows: batch idx + zero box
            float* o = out + ((size_t)b * POST_NMS_N + r) * 5;
            o[0] = (float)b; o[1] = o[2] = o[3] = o[4] = 0.f;
        }
    }
}

// ---------------------------------------------------------------------------
extern "C" void kernel_launch(void* const* d_in, const int* in_sizes, int n_in,
                              void* d_out, int out_size, void* d_ws, size_t ws_size,
                              hipStream_t stream) {
    const float* features = (const float*)d_in[0];
    const float* im_info  = (const float*)d_in[1];
    const float* W1 = (const float*)d_in[2];
    const float* b1 = (const float*)d_in[3];
    const float* Ws = (const float*)d_in[4];
    const float* bs = (const float*)d_in[5];
    const float* Wb = (const float*)d_in[6];
    const float* bb = (const float*)d_in[7];
    (void)in_sizes; (void)n_in; (void)out_size; (void)ws_size;

    char* ws = (char*)d_ws;
    size_t off = 0;
    auto carve = [&](size_t bytes) -> char* {
        char* p = ws + off;
        off = (off + bytes + 255) & ~(size_t)255;
        return p;
    };
    __bf16* Fb   = (__bf16*)carve((size_t)NB * NPIX * CIN * 2);   // 25.2 MB
    __bf16* Wt   = (__bf16*)carve((size_t)CMID * KTOT * 2);       //  7.1 MB
    __bf16* X    = (__bf16*)carve((size_t)MTOT * CMID * 2);       // 16.8 MB
    __bf16* Wh   = (__bf16*)carve((size_t)80 * CMID * 2);
    float*  Head = (float*)carve((size_t)MTOT * 80 * 4);          //  5.2 MB
    float*  Boxes= (float*)carve((size_t)NB * NA_TOT * 4 * 4);    //  3.1 MB
    unsigned long long* Keys = (unsigned long long*)carve((size_t)NB * SORTN * 8); // 2 MB
    float*  SB   = (float*)carve((size_t)NB * PRE_NMS_N * 4 * 4); //  0.4 MB

    prep_weights_km<<<(CMID * KTOT + 255) / 256, 256, 0, stream>>>(W1, Wt);
    prep_head_weights<<<(80 * CMID + 255) / 256, 256, 0, stream>>>(Ws, Wb, Wh);
    {
        size_t n = (size_t)NB * NPIX * CIN;
        prep_features_nhwc<<<(unsigned)((n + 255) / 256), 256, 0, stream>>>(features, Fb);
    }
    conv3x3_wmma<<<dim3(CMID / 128, MTOT / 128), 256, 0, stream>>>(Fb, Wt, b1, X);
    heads_wmma<<<MTOT / 128, 256, 0, stream>>>(X, Wh, Head);
    decode_kernel<<<(NB * SORTN + 255) / 256, 256, 0, stream>>>(Head, bs, bb, im_info, Boxes, Keys);
    for (int k = 2; k <= SORTN; k <<= 1)
        for (int j = k >> 1; j > 0; j >>= 1)
            bitonic_step<<<(NB * SORTN + 255) / 256, 256, 0, stream>>>(Keys, k, j);
    gather_sorted<<<(NB * PRE_NMS_N + 255) / 256, 256, 0, stream>>>(Keys, Boxes, SB);
    nms_kernel<<<NB, 1024, 0, stream>>>(SB, (float*)d_out);
}
